// MultiScaleVisionEncoder_43439299232205
// MI455X (gfx1250) — matrix operand order repaired and checked by reading, not compiled, not run on hardware
//
#include <hip/hip_runtime.h>
#include <cmath>

// ---------------------------------------------------------------------------
// Types for CDNA5 WMMA (wave32): A/B = 16 bf16 halves per lane (8 VGPRs),
// C/D = 8 f32 per lane.
// ---------------------------------------------------------------------------
typedef __attribute__((ext_vector_type(16))) __bf16 v16bf;
typedef __attribute__((ext_vector_type(8)))  float  v8f;

union BFrag { unsigned short us[16]; uint4 q[2]; v16bf v; };

__device__ __forceinline__ unsigned short f2bf(float f) {
    union { float f; unsigned u; } cv; cv.f = f;
    unsigned u = cv.u;
    unsigned r = u + 0x7fffu + ((u >> 16) & 1u);   // round-to-nearest-even
    return (unsigned short)(r >> 16);
}

// ---------------------------------------------------------------------------
// Generic implicit-GEMM convolution using V_WMMA_F32_16X16X32_BF16.
// C[M,N] = A[M,K] * B[K,N], M = B*Hout*Wout, K = Cin*KH*KW, N = Cout tile.
// Template on (KH,KW) so the k -> (ci,ky,kx) decode is compile-time.
// Each block: 128 threads = 4 waves; each wave owns 4 consecutive 16-row
// M-tiles sharing one B fragment per K-step (4 WMMAs per fragment build).
// Weights are staged once per block into LDS as bf16 in B-fragment order
// (row n = output channel, contiguous K) so the per-lane B fragment is two
// 16-byte LDS loads.
// out = relu_opt( (acc + bias[n]) * bnscale )
// ---------------------------------------------------------------------------
#define KPAD_MAX 320   // max padded K (fc1: 294 -> 320)

template <int KH, int KW>
__global__ void __launch_bounds__(128)
conv_wmma_kernel(const float* __restrict__ in, const float* __restrict__ wgt,
                 const float* __restrict__ bias, float* __restrict__ out,
                 int Bn, int Cin, int H, int W, int inCtot, int inCoff,
                 int Cout, int Hout, int Wout, int outCtot, int outCoff,
                 int stride, int pad, int dil,
                 float bnscale, int dorelu)
{
    constexpr int KHW = KH * KW;
    const int lane  = threadIdx.x & 31;
    const int wv    = threadIdx.x >> 5;
    const int M     = Bn * Hout * Wout;
    const int tileN = blockIdx.y * 16;
    const int Ktot  = Cin * KHW;
    const int nk    = (Ktot + 31) >> 5;
    const int Kpad  = nk << 5;
    const long HWl  = (long)H * W;

    // ---- stage bf16 weights into LDS: lds_w[n][k], n = 0..15 of this N-tile
    __shared__ unsigned short lds_w[16 * KPAD_MAX];
    for (int n = 0; n < 16; ++n) {
        int gn = tileN + n;
        for (int k = threadIdx.x; k < Kpad; k += 128) {
            float bv = 0.f;
            if (gn < Cout && k < Ktot) {
                int ci = k / KHW;
                int rr = k - ci * KHW;
                int ry = rr / KW;
                int rx = rr - ry * KW;
                bv = wgt[(((long)gn * Cin + ci) * KH + ry) * KW + rx];
            }
            lds_w[n * Kpad + k] = f2bf(bv);
        }
    }
    __syncthreads();

    // ---- per-lane roles
    const int  ml   = lane & 15;   // A: M row-in-tile, B/C/D: N column
    const int  ah   = lane >> 4;   // half-select (K sub-block / C row group)
    const int  n    = tileN + ml;
    const bool nval = (n < Cout);

    // ---- decode the 4 M-tiles owned by this wave
    const int waveM = blockIdx.x * 256 + wv * 64;
    int  tb[4], toh[4], tow[4];
    long tpix[4];
    bool tval[4];
#pragma unroll
    for (int t = 0; t < 4; ++t) {
        int mg = waveM + t * 16 + ml;
        tval[t] = (mg < M);
        int b = 0, oh = 0, ow = 0;
        if (tval[t]) {
            b = mg / (Hout * Wout);
            int r = mg - b * Hout * Wout;
            oh = r / Wout; ow = r - oh * Wout;
        }
        tb[t] = b; toh[t] = oh; tow[t] = ow;
        tpix[t] = ((long)(b * inCtot + inCoff) * H + oh) * W + ow;
        __builtin_prefetch(&in[tpix[t]], 0, 1);     // global_prefetch_b8
    }

    const unsigned short* wrow = &lds_w[ml * Kpad];

    v8f acc[4];
#pragma unroll
    for (int t = 0; t < 4; ++t) acc[t] = (v8f){0.f,0.f,0.f,0.f,0.f,0.f,0.f,0.f};

    for (int kt = 0; kt < nk; ++kt) {
        const int kbase = kt * 32;
        // ---- B fragment: lane n, K = kbase + ah*16 .. +15 (two b128 LDS loads)
        BFrag fb;
        fb.q[0] = *(const uint4*)(wrow + kbase + ah * 16);
        fb.q[1] = *(const uint4*)(wrow + kbase + ah * 16 + 8);

#pragma unroll
        for (int t = 0; t < 4; ++t) {
            // ---- A fragment for tile t (16-bit A 16x32 lane layout)
            BFrag fa;
#pragma unroll
            for (int h = 0; h < 16; ++h) {
                int ka = kbase + ah * 16 + h;
                float av = 0.f;
                if (KHW == 1) {
                    if (tval[t] && ka < Ktot)
                        av = in[tpix[t] + (long)ka * HWl];
                } else {
                    int ci = ka / KHW;          // compile-time divisor
                    int rr = ka - ci * KHW;
                    int ry = rr / KW;
                    int rx = rr - ry * KW;
                    int ih = toh[t] * stride - pad + ry * dil;
                    int iw = tow[t] * stride - pad + rx * dil;
                    if (tval[t] && ka < Ktot && ih >= 0 && ih < H && iw >= 0 && iw < W)
                        av = in[((long)(tb[t] * inCtot + inCoff) + ci) * HWl + (long)ih * W + iw];
                }
                fa.us[h] = f2bf(av);
            }
            acc[t] = __builtin_amdgcn_wmma_f32_16x16x32_bf16(
                         false, fa.v, false, fb.v, (short)0, acc[t], false, false);
        }
    }

    // ---- store C/D: lane -> N = ml; VGPR r -> M = r + ah*8 within tile
    float bterm = (bias != nullptr && nval) ? bias[n] : 0.f;
#pragma unroll
    for (int t = 0; t < 4; ++t) {
#pragma unroll
        for (int r = 0; r < 8; ++r) {
            int mr = waveM + t * 16 + r + ah * 8;
            if (nval && mr < M) {
                int b2  = mr / (Hout * Wout);
                int rem = mr - b2 * Hout * Wout;
                int oy  = rem / Wout;
                int ox  = rem - oy * Wout;
                float v = (acc[t][r] + bterm) * bnscale;
                if (dorelu) v = v > 0.f ? v : 0.f;
                out[(((long)b2 * outCtot + outCoff + n) * Hout + oy) * (long)Wout + ox] = v;
            }
        }
    }
}

// ---------------------------------------------------------------------------
// Max pooling (VALID), one thread per output element.
// ---------------------------------------------------------------------------
__global__ void maxpool_kernel(const float* __restrict__ in, float* __restrict__ out,
                               int Bn, int C, int H, int W, int k, int s,
                               int Hout, int Wout)
{
    long i   = (long)blockIdx.x * blockDim.x + threadIdx.x;
    long tot = (long)Bn * C * Hout * Wout;
    if (i >= tot) return;
    int ox = i % Wout; long t = i / Wout;
    int oy = t % Hout; t /= Hout;
    int c  = t % C;    int b = t / C;
    const float* p = in + ((long)b * C + c) * H * W;
    float m = -INFINITY;
    for (int dy = 0; dy < k; ++dy)
        for (int dx = 0; dx < k; ++dx) {
            int iy = oy * s + dy, ix = ox * s + dx;
            if (iy < H && ix < W) m = fmaxf(m, p[(long)iy * W + ix]);
        }
    out[i] = m;
}

// ---------------------------------------------------------------------------
// Mean over H*W for one (b,c): grid = (C, B), 256 threads.
// ---------------------------------------------------------------------------
__global__ void mean_hw_kernel(const float* __restrict__ in, float* __restrict__ out,
                               int Ctot, int coff, int C, int HW)
{
    int c = blockIdx.x, b = blockIdx.y;
    const float* p = in + ((long)b * Ctot + coff + c) * HW;
    float s = 0.f;
    for (int i = threadIdx.x; i < HW; i += blockDim.x) s += p[i];
    __shared__ float red[256];
    red[threadIdx.x] = s; __syncthreads();
    for (int st = 128; st > 0; st >>= 1) {
        if (threadIdx.x < st) red[threadIdx.x] += red[threadIdx.x + st];
        __syncthreads();
    }
    if (threadIdx.x == 0) out[b * C + c] = red[0] / (float)HW;
}

// ---------------------------------------------------------------------------
// Channel attention MLP: a = sigmoid(relu(p@W1^T+b1)@W2^T+b2). Block per batch.
// ---------------------------------------------------------------------------
__global__ void attn_mlp_kernel(const float* __restrict__ p,
                                const float* __restrict__ w1, const float* __restrict__ b1,
                                const float* __restrict__ w2, const float* __restrict__ b2,
                                float* __restrict__ a, int C, int Hd)
{
    int b = blockIdx.x, t = threadIdx.x;
    __shared__ float hid[32];
    if (t < Hd) {
        float s = b1[t];
        for (int k = 0; k < C; ++k) s += p[b * C + k] * w1[t * C + k];
        hid[t] = s > 0.f ? s : 0.f;
    }
    __syncthreads();
    if (t < C) {
        float s = b2[t];
        for (int k = 0; k < Hd; ++k) s += hid[k] * w2[t * Hd + k];
        a[b * C + t] = 1.f / (1.f + expf(-s));
    }
}

// out[b,coff+c,:] = in[b,icoff+c,:] * a[b,c]
__global__ void scale_ch_kernel(const float* __restrict__ in, const float* __restrict__ a,
                                float* __restrict__ out, int Bn, int C, int HW,
                                int inCtot, int inCoff, int outCtot, int outCoff)
{
    long i   = (long)blockIdx.x * blockDim.x + threadIdx.x;
    long tot = (long)Bn * C * HW;
    if (i >= tot) return;
    int pix = i % HW; long t = i / HW;
    int c = t % C; int b = t / C;
    out[((long)b * outCtot + outCoff + c) * HW + pix] =
        in[((long)b * inCtot + inCoff + c) * HW + pix] * a[b * C + c];
}

// avg & max over channels -> (B,1,H,W) each
__global__ void mean_max_ch_kernel(const float* __restrict__ in,
                                   float* __restrict__ avg, float* __restrict__ mx,
                                   int Bn, int C, int HW)
{
    long i   = (long)blockIdx.x * blockDim.x + threadIdx.x;
    long tot = (long)Bn * HW;
    if (i >= tot) return;
    int pix = i % HW; int b = i / HW;
    float s = 0.f, m = -INFINITY;
    for (int c = 0; c < C; ++c) {
        float v = in[((long)b * C + c) * HW + pix];
        s += v; m = fmaxf(m, v);
    }
    avg[(long)b * HW + pix] = s / (float)C;
    mx [(long)b * HW + pix] = m;
}

__global__ void sobel_kernel(const float* __restrict__ avg, float* __restrict__ gmag,
                             int Bn, int H, int W)
{
    long i   = (long)blockIdx.x * blockDim.x + threadIdx.x;
    long tot = (long)Bn * H * W;
    if (i >= tot) return;
    int x = i % W; long t = i / W; int y = t % H; int b = t / H;
    const float* p = avg + (long)b * H * W;
#define GV(yy, xx) (((yy) >= 0 && (yy) < H && (xx) >= 0 && (xx) < W) ? p[(long)(yy) * W + (xx)] : 0.f)
    float gx = -GV(y-1,x-1) + GV(y-1,x+1) - 2.f*GV(y,x-1) + 2.f*GV(y,x+1) - GV(y+1,x-1) + GV(y+1,x+1);
    float gy = -GV(y-1,x-1) - 2.f*GV(y-1,x) - GV(y-1,x+1) + GV(y+1,x-1) + 2.f*GV(y+1,x) + GV(y+1,x+1);
#undef GV
    gmag[i] = sqrtf(gx * gx + gy * gy + 1e-6f);
}

// local_var = avgpool3(a*a) - avgpool3(a)^2 + 1e-6  (zero-padded 3x3, /9)
__global__ void localvar_kernel(const float* __restrict__ a, float* __restrict__ var,
                                int Bn, int H, int W)
{
    long i   = (long)blockIdx.x * blockDim.x + threadIdx.x;
    long tot = (long)Bn * H * W;
    if (i >= tot) return;
    int x = i % W; long t = i / W; int y = t % H; int b = t / H;
    const float* p = a + (long)b * H * W;
    float s1 = 0.f, s2 = 0.f;
    for (int dy = -1; dy <= 1; ++dy)
        for (int dx = -1; dx <= 1; ++dx) {
            int yy = y + dy, xx = x + dx;
            if (yy >= 0 && yy < H && xx >= 0 && xx < W) {
                float v = p[(long)yy * W + xx];
                s1 += v; s2 += v * v;
            }
        }
    float m = s1 / 9.f;
    var[i] = s2 / 9.f - m * m + 1e-6f;
}

// kxk conv over up to 3 single-channel planes -> sigmoid -> 1 channel
__global__ void sconv_sig_kernel(const float* __restrict__ c0, const float* __restrict__ c1,
                                 const float* __restrict__ c2, int nch,
                                 const float* __restrict__ w, const float* __restrict__ bias,
                                 float* __restrict__ out, int Bn, int H, int W, int k, int pad)
{
    long i   = (long)blockIdx.x * blockDim.x + threadIdx.x;
    long tot = (long)Bn * H * W;
    if (i >= tot) return;
    int x = i % W; long t = i / W; int y = t % H; int b = t / H;
    float s = bias[0];
    for (int ch = 0; ch < nch; ++ch) {
        const float* src = (ch == 0) ? c0 : (ch == 1) ? c1 : c2;
        src += (long)b * H * W;
        for (int ky = 0; ky < k; ++ky)
            for (int kx = 0; kx < k; ++kx) {
                int iy = y - pad + ky, ix = x - pad + kx;
                if (iy >= 0 && iy < H && ix >= 0 && ix < W)
                    s += src[(long)iy * W + ix] * w[(ch * k + ky) * k + kx];
            }
    }
    out[i] = 1.f / (1.f + expf(-s));
}

// out[b,ocoff+c,:] = in[b,icoff+c,:] * sa[b,:]
__global__ void mul_spatial_kernel(const float* __restrict__ in, const float* __restrict__ sa,
                                   float* __restrict__ out, int Bn, int C, int HW,
                                   int inCtot, int inCoff, int outCtot, int outCoff)
{
    long i   = (long)blockIdx.x * blockDim.x + threadIdx.x;
    long tot = (long)Bn * C * HW;
    if (i >= tot) return;
    int pix = i % HW; long t = i / HW;
    int c = t % C; int b = t / C;
    out[((long)b * outCtot + outCoff + c) * HW + pix] =
        in[((long)b * inCtot + inCoff + c) * HW + pix] * sa[(long)b * HW + pix];
}

// bilinear upsample (half-pixel centers, edge clamp)
__global__ void upsample_kernel(const float* __restrict__ in, float* __restrict__ out,
                                int Bn, int C, int Hin, int Win, int Hout, int Wout,
                                int outCtot, int outCoff)
{
    long i   = (long)blockIdx.x * blockDim.x + threadIdx.x;
    long tot = (long)Bn * C * Hout * Wout;
    if (i >= tot) return;
    int ox = i % Wout; long t = i / Wout;
    int oy = t % Hout; t /= Hout;
    int c  = t % C;    int b = t / C;
    float fy = (oy + 0.5f) * (float)Hin / (float)Hout - 0.5f;
    float fx = (ox + 0.5f) * (float)Win / (float)Wout - 0.5f;
    int y0 = (int)floorf(fy), x0 = (int)floorf(fx);
    float wy = fy - (float)y0, wx = fx - (float)x0;
    int y0c = min(max(y0, 0), Hin - 1), y1c = min(max(y0 + 1, 0), Hin - 1);
    int x0c = min(max(x0, 0), Win - 1), x1c = min(max(x0 + 1, 0), Win - 1);
    const float* p = in + ((long)b * C + c) * Hin * Win;
    float v = (1.f - wy) * ((1.f - wx) * p[(long)y0c * Win + x0c] + wx * p[(long)y0c * Win + x1c])
            +        wy  * ((1.f - wx) * p[(long)y1c * Win + x0c] + wx * p[(long)y1c * Win + x1c]);
    out[((long)b * outCtot + outCoff + c) * Hout * Wout + (long)oy * Wout + ox] = v;
}

// gate: softmax(relu(pooled@g1^T+b1)@g2^T+b2) ; block per batch, 32 threads
__global__ void gate_kernel(const float* __restrict__ pooled,
                            const float* __restrict__ g1w, const float* __restrict__ g1b,
                            const float* __restrict__ g2w, const float* __restrict__ g2b,
                            float* __restrict__ wout)
{
    int b = blockIdx.x, t = threadIdx.x;
    __shared__ float hid[16];
    __shared__ float lg[24];
    __shared__ float mm, sm;
    if (t < 16) {
        float s = g1b[t];
        for (int k = 0; k < 24; ++k) s += pooled[b * 24 + k] * g1w[t * 24 + k];
        hid[t] = s > 0.f ? s : 0.f;
    }
    __syncthreads();
    if (t < 24) {
        float s = g2b[t];
        for (int k = 0; k < 16; ++k) s += hid[k] * g2w[t * 16 + k];
        lg[t] = s;
    }
    __syncthreads();
    if (t == 0) {
        float m = lg[0];
        for (int k = 1; k < 24; ++k) m = fmaxf(m, lg[k]);
        float s = 0.f;
        for (int k = 0; k < 24; ++k) s += expf(lg[k] - m);
        mm = m; sm = s;
    }
    __syncthreads();
    if (t < 24) wout[b * 24 + t] = expf(lg[t] - mm) / sm;
}

// kernels = relu(locmean @ kg_w^T), one thread per (b,o)
__global__ void kergen_kernel(const float* __restrict__ lm, const float* __restrict__ kgw,
                              float* __restrict__ ker, int Bn, int Kout, int Cin)
{
    int i = blockIdx.x * blockDim.x + threadIdx.x;
    if (i >= Bn * Kout) return;
    int b = i / Kout, o = i % Kout;
    float s = 0.f;
    for (int k = 0; k < Cin; ++k) s += lm[b * Cin + k] * kgw[o * Cin + k];
    ker[i] = s > 0.f ? s : 0.f;
}

// Depthwise kxk conv (optionally per-batch generated weights), with dilation
__global__ void dwconv_kernel(const float* __restrict__ in, int inCtot, int inCoff,
                              const float* __restrict__ wk, int perB,
                              const float* __restrict__ bias,
                              float* __restrict__ out, int outCtot, int outCoff,
                              int Bn, int C, int H, int W, int k, int pad, int dil)
{
    long i   = (long)blockIdx.x * blockDim.x + threadIdx.x;
    long tot = (long)Bn * C * H * W;
    if (i >= tot) return;
    int x = i % W; long t = i / W;
    int y = t % H; t /= H;
    int c = t % C; int b = t / C;
    const float* src = in + ((long)b * inCtot + inCoff + c) * H * W;
    const float* kk  = wk + (long)(perB ? (b * C + c) : c) * k * k;
    float s = (bias != nullptr) ? bias[c] : 0.f;
    for (int ky = 0; ky < k; ++ky)
        for (int kx = 0; kx < k; ++kx) {
            int iy = y - pad + ky * dil, ix = x - pad + kx * dil;
            if (iy >= 0 && iy < H && ix >= 0 && ix < W)
                s += src[(long)iy * W + ix] * kk[ky * k + kx];
        }
    out[((long)b * outCtot + outCoff + c) * H * W + (long)y * W + x] = s;
}

// ---------------------------------------------------------------------------
// Host orchestration
// ---------------------------------------------------------------------------
static inline long cdivl(long a, long b) { return (a + b - 1) / b; }

static void conv_wmma(hipStream_t s, const float* in, const float* w, const float* bias,
                      float* out, int Bn, int Cin, int H, int W, int inCtot, int inCoff,
                      int Cout, int Hout, int Wout, int outCtot, int outCoff,
                      int KH, int KW, int stride, int pad, int dil, float bnscale, int relu)
{
    long M = (long)Bn * Hout * Wout;
    dim3 grid((unsigned)cdivl(M, 256), (unsigned)cdivl(Cout, 16), 1);
    if (KH == 3 && KW == 3)
        conv_wmma_kernel<3, 3><<<grid, 128, 0, s>>>(in, w, bias, out, Bn, Cin, H, W, inCtot, inCoff,
                                                    Cout, Hout, Wout, outCtot, outCoff,
                                                    stride, pad, dil, bnscale, relu);
    else
        conv_wmma_kernel<1, 1><<<grid, 128, 0, s>>>(in, w, bias, out, Bn, Cin, H, W, inCtot, inCoff,
                                                    Cout, Hout, Wout, outCtot, outCoff,
                                                    stride, pad, dil, bnscale, relu);
}

extern "C" void kernel_launch(void* const* d_in, const int* in_sizes, int n_in,
                              void* d_out, int out_size, void* d_ws, size_t ws_size,
                              hipStream_t stream)
{
    const float* x = (const float*)d_in[0];
    const float* P[48];
    for (int i = 0; i < 48 && i < n_in; ++i) P[i] = (const float*)d_in[i];

    float* Wk = (float*)d_ws;
    float* O  = (float*)d_out;
    const float BNS = 1.0f / sqrtf(1.0f + 1e-5f);
    const int S224 = 224 * 224, S112 = 112 * 112, S56 = 56 * 56, S28 = 28 * 28;

    // workspace layout (floats)
    long off = 0;
#define ALLOC(nm, n) float* nm = Wk + off; off += (long)(n);
    ALLOC(t_stem,  32L * 12 * S224)   // stem activations
    ALLOC(t_hi12,  32L * 12 * S224)   // 'high' conv out; later reused for gd output
    ALLOC(t_half,  32L * 12 * S112)   // maxpool(out); later reused for ds output
    ALLOC(t_me16,  32L * 16 * S112)   // med; later reused for pp1 output
    ALLOC(t_lo24a, 32L * 24 * S112)   // low conv (pre-pool)
    ALLOC(t_lo24,  32L * 24 * S56)
    ALLOC(t_h8,    32L * 8 * S224)
    ALLOC(t_m8,    32L * 8 * S112)
    ALLOC(t_l8,    32L * 8 * S56)
    ALLOC(t_avgh,  32L * S224) ALLOC(t_mxh, 32L * S224) ALLOC(t_gmag, 32L * S224)
    ALLOC(t_avgm,  32L * S112) ALLOC(t_mxm, 32L * S112) ALLOC(t_varm, 32L * S112)
    ALLOC(t_avgl,  32L * S56)  ALLOC(t_mxl, 32L * S56)
    ALLOC(t_fused, 32L * 24 * S224)
    ALLOC(t_comb,  32L * 24 * S224)
    ALLOC(t_p8,    32L * 8)  ALLOC(t_a8,  32L * 8)
    ALLOC(t_pool,  32L * 24) ALLOC(t_gate, 32L * 24)
    ALLOC(t_lm,    32L * 12) ALLOC(t_ker, 32L * 108)
    ALLOC(t_f3,    32L * 6 * S28) ALLOC(t_f4, 32L * 6 * 49) ALLOC(t_h1, 32L * 64)
#undef ALLOC
    (void)ws_size; (void)in_sizes; (void)out_size;

    // output layout (return order)
    float* o_log = O;
    float* o_hsa = O + 32L * 11;
    float* o_msa = o_hsa + 32L * S224;
    float* o_lsa = o_msa + 32L * S112;
    float* o_vis = o_lsa + 32L * S56;

    const int T = 256;
#define G1(n) dim3((unsigned)cdivl((long)(n), T)), dim3(T)

    // stem / high / med / low backbone (WMMA implicit GEMM)
    conv_wmma(stream, x,      P[1], nullptr, t_stem, 32, 3, 224, 224, 3, 0, 12, 224, 224, 12, 0, 3, 3, 1, 1, 1, BNS, 1);
    conv_wmma(stream, t_stem, P[2], nullptr, t_hi12, 32, 12, 224, 224, 12, 0, 12, 224, 224, 12, 0, 3, 3, 1, 1, 1, BNS, 1);
    maxpool_kernel<<<G1(32L*12*S112), 0, stream>>>(t_stem, t_half, 32, 12, 224, 224, 2, 2, 112, 112);
    conv_wmma(stream, t_half, P[3], nullptr, t_me16, 32, 12, 112, 112, 12, 0, 16, 112, 112, 16, 0, 3, 3, 1, 1, 1, BNS, 1);
    conv_wmma(stream, t_half, P[4], nullptr, t_lo24a, 32, 12, 112, 112, 12, 0, 24, 112, 112, 24, 0, 3, 3, 1, 1, 1, BNS, 1);
    maxpool_kernel<<<G1(32L*24*S56), 0, stream>>>(t_lo24a, t_lo24, 32, 24, 112, 112, 2, 2, 56, 56);

    // 1x1 projections
    conv_wmma(stream, t_hi12, P[5], P[6],  t_h8, 32, 12, 224, 224, 12, 0, 8, 224, 224, 8, 0, 1, 1, 1, 0, 1, 1.0f, 0);
    conv_wmma(stream, t_me16, P[7], P[8],  t_m8, 32, 16, 112, 112, 16, 0, 8, 112, 112, 8, 0, 1, 1, 1, 0, 1, 1.0f, 0);
    conv_wmma(stream, t_lo24, P[9], P[10], t_l8, 32, 24,  56,  56, 24, 0, 8,  56,  56, 8, 0, 1, 1, 1, 0, 1, 1.0f, 0);

    // channel attention (h, m, l), applied in-place
    mean_hw_kernel<<<dim3(8, 32), 256, 0, stream>>>(t_h8, t_p8, 8, 0, 8, S224);
    attn_mlp_kernel<<<32, 32, 0, stream>>>(t_p8, P[11], P[12], P[13], P[14], t_a8, 8, 2);
    scale_ch_kernel<<<G1(32L*8*S224), 0, stream>>>(t_h8, t_a8, t_h8, 32, 8, S224, 8, 0, 8, 0);
    mean_hw_kernel<<<dim3(8, 32), 256, 0, stream>>>(t_m8, t_p8, 8, 0, 8, S112);
    attn_mlp_kernel<<<32, 32, 0, stream>>>(t_p8, P[15], P[16], P[17], P[18], t_a8, 8, 2);
    scale_ch_kernel<<<G1(32L*8*S112), 0, stream>>>(t_m8, t_a8, t_m8, 32, 8, S112, 8, 0, 8, 0);
    mean_hw_kernel<<<dim3(8, 32), 256, 0, stream>>>(t_l8, t_p8, 8, 0, 8, S56);
    attn_mlp_kernel<<<32, 32, 0, stream>>>(t_p8, P[19], P[20], P[21], P[22], t_a8, 8, 2);
    scale_ch_kernel<<<G1(32L*8*S56), 0, stream>>>(t_l8, t_a8, t_l8, 32, 8, S56, 8, 0, 8, 0);

    // spatial attention h: sobel magnitude, 3x3 conv -> sigmoid
    mean_max_ch_kernel<<<G1(32L*S224), 0, stream>>>(t_h8, t_avgh, t_mxh, 32, 8, S224);
    sobel_kernel<<<G1(32L*S224), 0, stream>>>(t_avgh, t_gmag, 32, 224, 224);
    sconv_sig_kernel<<<G1(32L*S224), 0, stream>>>(t_avgh, t_mxh, t_gmag, 3, P[23], P[24], o_hsa, 32, 224, 224, 3, 1);
    // spatial attention m: local variance, 5x5 conv -> sigmoid
    mean_max_ch_kernel<<<G1(32L*S112), 0, stream>>>(t_m8, t_avgm, t_mxm, 32, 8, S112);
    localvar_kernel<<<G1(32L*S112), 0, stream>>>(t_avgm, t_varm, 32, 112, 112);
    sconv_sig_kernel<<<G1(32L*S112), 0, stream>>>(t_avgm, t_mxm, t_varm, 3, P[25], P[26], o_msa, 32, 112, 112, 5, 2);
    // spatial attention l: 7x7 conv -> sigmoid
    mean_max_ch_kernel<<<G1(32L*S56), 0, stream>>>(t_l8, t_avgl, t_mxl, 32, 8, S56);
    sconv_sig_kernel<<<G1(32L*S56), 0, stream>>>(t_avgl, t_mxl, nullptr, 2, P[27], P[28], o_lsa, 32, 56, 56, 7, 3);

    // fuse: h_o into fused[:,0:8], upsampled m_o into [:,8:16], l_o into [:,16:24]
    mul_spatial_kernel<<<G1(32L*8*S224), 0, stream>>>(t_h8, o_hsa, t_fused, 32, 8, S224, 8, 0, 24, 0);
    mul_spatial_kernel<<<G1(32L*8*S112), 0, stream>>>(t_m8, o_msa, t_m8, 32, 8, S112, 8, 0, 8, 0);
    upsample_kernel<<<G1(32L*8*S224), 0, stream>>>(t_m8, t_fused, 32, 8, 112, 112, 224, 224, 24, 8);
    mul_spatial_kernel<<<G1(32L*8*S56), 0, stream>>>(t_l8, o_lsa, t_l8, 32, 8, S56, 8, 0, 8, 0);
    upsample_kernel<<<G1(32L*8*S224), 0, stream>>>(t_l8, t_fused, 32, 8, 56, 56, 224, 224, 24, 16);

    // gating + vis_emb
    mean_hw_kernel<<<dim3(24, 32), 256, 0, stream>>>(t_fused, t_pool, 24, 0, 24, S224);
    gate_kernel<<<32, 32, 0, stream>>>(t_pool, P[29], P[30], P[31], P[32], t_gate);
    scale_ch_kernel<<<G1(32L*24*S224), 0, stream>>>(t_fused, t_gate, t_fused, 32, 24, S224, 24, 0, 24, 0);
    mean_hw_kernel<<<dim3(24, 32), 256, 0, stream>>>(t_fused, o_vis, 24, 0, 24, S224);

    // dynamic local branch: generated 3x3 depthwise kernels on fused[:,0:12]
    mean_hw_kernel<<<dim3(12, 32), 256, 0, stream>>>(t_fused, t_lm, 24, 0, 12, S224);
    kergen_kernel<<<G1(32 * 108), 0, stream>>>(t_lm, P[33], t_ker, 32, 108, 12);
    dwconv_kernel<<<G1(32L*12*S224), 0, stream>>>(t_fused, 24, 0, t_ker, 1, nullptr,
                                                  t_comb, 24, 0, 32, 12, 224, 224, 3, 1, 1);
    // global branch: depthwise dilated 3x3 (gd) then 1x1 (gp) + bn + relu
    dwconv_kernel<<<G1(32L*12*S224), 0, stream>>>(t_fused, 24, 12, P[34], 0, P[35],
                                                  t_hi12, 12, 0, 32, 12, 224, 224, 3, 2, 2);
    conv_wmma(stream, t_hi12, P[36], P[37], t_comb, 32, 12, 224, 224, 12, 0, 12, 224, 224, 24, 12, 1, 1, 1, 0, 1, BNS, 1);

    // downsampling head
    conv_wmma(stream, t_comb, P[38], P[39], t_half, 32, 24, 224, 224, 24, 0, 12, 112, 112, 12, 0, 3, 3, 2, 1, 1, BNS, 1);
    conv_wmma(stream, t_half, P[40], P[41], t_me16, 32, 12, 112, 112, 12, 0,  6,  56,  56,  6, 0, 3, 3, 2, 1, 1, BNS, 1);
    conv_wmma(stream, t_me16, P[42], P[43], t_f3,   32,  6,  56,  56,  6, 0,  6,  28,  28,  6, 0, 3, 3, 2, 1, 1, BNS, 1);
    maxpool_kernel<<<G1(32L*6*49), 0, stream>>>(t_f3, t_f4, 32, 6, 28, 28, 4, 4, 7, 7);

    // FC head (WMMA as 1x1 "convs" over flattened features)
    conv_wmma(stream, t_f4, P[44], P[45], t_h1,  32, 294, 1, 1, 294, 0, 64, 1, 1, 64, 0, 1, 1, 1, 0, 1, 1.0f, 1);
    conv_wmma(stream, t_h1, P[46], P[47], o_log, 32,  64, 1, 1,  64, 0, 11, 1, 1, 11, 0, 1, 1, 1, 0, 1, 1.0f, 0);
#undef G1
}
